// GCNLayer_68315749810546
// MI455X (gfx1250) — matrix operand order repaired and checked by reading, not compiled, not run on hardware
//
#include <hip/hip_runtime.h>
#include <hip/hip_bf16.h>
#include <math.h>

#define NODES 661
#define FDIM  64

typedef __attribute__((ext_vector_type(16))) _Float16 v16h;
typedef __attribute__((ext_vector_type(8)))  _Float16 v8h;
typedef __attribute__((ext_vector_type(8)))  float    v8f;

// ---------------------------------------------------------------------------
// SPMM: one wave per edge, 2 features per lane, fused over both e and f.
// out[r] += v * x[c]  (unsafeAtomicAdd -> global_atomic_add_f32)
// ---------------------------------------------------------------------------
__global__ __launch_bounds__(256)
void spmm2_kernel(const int* __restrict__ rows, const int* __restrict__ cols,
                  const float* __restrict__ vals,
                  const float* __restrict__ xe, const float* __restrict__ xf,
                  float* __restrict__ oe, float* __restrict__ of, int nnz)
{
    const int lane = threadIdx.x & 31;
    const int edge = blockIdx.x * 8 + (threadIdx.x >> 5);
    if (edge >= nnz) return;
    const int   r = rows[edge];
    const int   c = cols[edge];
    const float v = vals[edge];
    const int  fo = lane * 2;
    const float2 ev = *(const float2*)(xe + (size_t)c * FDIM + fo);
    const float2 fv = *(const float2*)(xf + (size_t)c * FDIM + fo);
    float* pe = oe + (size_t)r * FDIM + fo;
    float* pf = of + (size_t)r * FDIM + fo;
    unsafeAtomicAdd(pe + 0, v * ev.x);
    unsafeAtomicAdd(pe + 1, v * ev.y);
    unsafeAtomicAdd(pf + 0, v * fv.x);
    unsafeAtomicAdd(pf + 1, v * fv.y);
}

// ---------------------------------------------------------------------------
// Elementwise algebra. Buffers updated IN PLACE:
//   b0: e_G -> e3,  b1: f_G -> new_e,  b2: e_B -> f3,  b3: f_B -> new_f
// ---------------------------------------------------------------------------
__global__ __launch_bounds__(256)
void elemwise_kernel(const float* __restrict__ e, const float* __restrict__ f,
                     float* __restrict__ b0, float* __restrict__ b1,
                     float* __restrict__ b2, float* __restrict__ b3,
                     const float* __restrict__ Gd, const float* __restrict__ Bd,
                     const float* __restrict__ Pd, const float* __restrict__ Qd,
                     long total)
{
    long i = (long)blockIdx.x * blockDim.x + threadIdx.x;
    if (i >= total) return;
    const long row = i >> 6;

    const float ev = e[i], fv = f[i];
    const float eG = b0[i], fG = b1[i], eB = b2[i], fB = b3[i];
    const float G = Gd[row], B = Bd[row], P = Pd[row], Q = Qd[row];

    const float base   = ev * ev + fv * fv + 0.1f;
    const float rbase  = 1.0f / base;
    const float alpha  = (P * ev + Q * fv) * rbase - eG - fB;
    const float beta   = (Q * ev - P * fv) * rbase + fG + eB;
    const float rgb    = 1.0f / (G * G + B * B);
    const float e3     = (alpha * G + beta * B) * rgb;
    const float f3     = (beta * G - alpha * B) * rgb;
    const float base1  = eG - fB;
    const float base2  = fG + eB;
    const float v2     = ev * ev + fv * fv;
    const float P_     = P - v2 * G;
    const float Q_     = Q + v2 * B;
    const float new_e  = (P_ * base1 + Q_ * base2) * rgb;
    const float new_f  = (P_ * base2 - Q_ * base1) * rgb;

    b0[i] = e3;
    b1[i] = new_e;
    b2[i] = f3;
    b3[i] = new_f;
}

// ---------------------------------------------------------------------------
// Attention: one block per graph; 8 pooled sigmoid scalars, normalized in
// groups of 4; written as [4][nGraphs] per side.
// ---------------------------------------------------------------------------
__global__ __launch_bounds__(256)
void attention_kernel(const float* __restrict__ m0, const float* __restrict__ m1,
                      const float* __restrict__ m2, const float* __restrict__ m3,
                      const float* __restrict__ m4, const float* __restrict__ m5,
                      const float* __restrict__ m6, const float* __restrict__ m7,
                      const float* __restrict__ w_e, const float* __restrict__ b_e,
                      const float* __restrict__ w_f, const float* __restrict__ b_f,
                      float* __restrict__ att_e, float* __restrict__ att_f,
                      int nGraphs)
{
    __shared__ float sacc[8];
    const int tid = threadIdx.x;
    if (tid < 8) sacc[tid] = 0.0f;
    __syncthreads();

    const int g = blockIdx.x;
    const size_t base = (size_t)g * NODES * FDIM;
    float acc[8] = {0.f, 0.f, 0.f, 0.f, 0.f, 0.f, 0.f, 0.f};

    for (int idx = tid; idx < NODES * FDIM; idx += 256) {
        const int fe = idx & (FDIM - 1);
        const float we = w_e[fe];
        const float wf = w_f[fe];
        const size_t k = base + idx;
        acc[0] += m0[k] * we;  acc[1] += m1[k] * we;
        acc[2] += m2[k] * we;  acc[3] += m3[k] * we;
        acc[4] += m4[k] * wf;  acc[5] += m5[k] * wf;
        acc[6] += m6[k] * wf;  acc[7] += m7[k] * wf;
    }
    #pragma unroll
    for (int j = 0; j < 8; ++j) {
        float v = acc[j];
        #pragma unroll
        for (int o = 16; o > 0; o >>= 1) v += __shfl_xor(v, o, 32);
        if ((tid & 31) == 0) atomicAdd(&sacc[j], v);
    }
    __syncthreads();

    if (tid == 0) {
        const float inv_nodes = 1.0f / (float)NODES;
        float se[4], sf[4];
        float de = 1e-4f, df = 1e-4f;
        #pragma unroll
        for (int j = 0; j < 4; ++j) {
            se[j] = 1.0f / (1.0f + expf(-(sacc[j]     * inv_nodes + b_e[0])));
            sf[j] = 1.0f / (1.0f + expf(-(sacc[4 + j] * inv_nodes + b_f[0])));
            de += se[j];
            df += sf[j];
        }
        #pragma unroll
        for (int j = 0; j < 4; ++j) {
            att_e[j * nGraphs + g] = se[j] / de;
            att_f[j * nGraphs + g] = sf[j] / df;
        }
    }
}

// ---------------------------------------------------------------------------
// Fused concat-GEMM + bias + tanh using v_wmma_f32_16x16x32_f16.
//   out[n,64] = tanh( [a0*X0 | a1*X1 | a2*X2 | a3*X3 | X4] @ W^T + bias )
//
// Block = 256 threads = 8 waves; wave w owns rows [blk*128 + 16w, +16) and
// computes ALL four 16-col N-tiles (A loaded exactly once per row; B shared
// from LDS). K = 320 -> 10 K-tiles x 4 N-tiles = 40 WMMA per wave.
//
// Fragment layouts (ISA 7.12.2):
//   A 16x32 f16: lane L holds row (L&15);
//     lanes 0-15:  elems 0-7 -> K 0..7,  elems 8-15 -> K 16..23
//     lanes 16-31: elems 0-7 -> K 8..15, elems 8-15 -> K 24..31
//   B 32x16: lane L holds col (L&15); elems 0-7 -> K 8*(L>>4)+0..7, elems 8-15 -> K 16+8*(L>>4)+0..7.
//   C/D 16x16 f32: lane L holds col (L&15); VGPR v -> row 8*(L>>4) + v.
// ---------------------------------------------------------------------------
__global__ __launch_bounds__(256)
void gemm_att_tanh_kernel(const float* __restrict__ X0, const float* __restrict__ X1,
                          const float* __restrict__ X2, const float* __restrict__ X3,
                          const float* __restrict__ X4,
                          const float* __restrict__ att,   // [4][nGraphs]
                          const float* __restrict__ W,     // [64][320] row-major
                          const float* __restrict__ bias,  // [64]
                          float* __restrict__ out,         // [n][64]
                          int n, int nGraphs)
{
    __shared__ __align__(16) _Float16 sW[64 * 320];
    const int tid = threadIdx.x;
    // Stage weights into LDS as f16 once per block.
    for (int i = tid; i < 64 * 320; i += 256) sW[i] = (_Float16)W[i];
    __syncthreads();

    const int lane  = tid & 31;
    const int wave  = tid >> 5;
    const int mbase = blockIdx.x * 128 + wave * 16;
    int row = mbase + (lane & 15);
    if (row >= n) row = n - 1;                 // clamp (stores bounds-checked)
    const int g   = row / NODES;
    const int hi  = lane >> 4;                 // 0 or 1 (lane half)
    const int bk  = hi * 8;                    // A-fragment K sub-block base
    const int nc  = lane & 15;                 // column-within-N-tile

    const float* Xs[5] = {X0, X1, X2, X3, X4};
    float scales[5];
    scales[0] = att[0 * nGraphs + g];
    scales[1] = att[1 * nGraphs + g];
    scales[2] = att[2 * nGraphs + g];
    scales[3] = att[3 * nGraphs + g];
    scales[4] = 1.0f;

    v8f c0 = {}, c1 = {}, c2 = {}, c3 = {};
    #pragma unroll
    for (int kt = 0; kt < 10; ++kt) {
        const int s  = kt >> 1;                // concat segment (64-wide)
        const int fb = (kt & 1) * 32;          // feature base inside segment
        const float* xp = Xs[s] + (size_t)row * FDIM + fb + bk;

        if (kt < 9) {                          // prefetch next K-tile A source
            const int s2  = (kt + 1) >> 1;
            const int fb2 = ((kt + 1) & 1) * 32;
            __builtin_prefetch(Xs[s2] + (size_t)row * FDIM + fb2 + bk, 0, 3);
        }

        const float4 p0 = *(const float4*)(xp + 0);
        const float4 p1 = *(const float4*)(xp + 4);
        const float4 p2 = *(const float4*)(xp + 16);
        const float4 p3 = *(const float4*)(xp + 20);
        const float sc = scales[s];

        v16h a;
        a[0]  = (_Float16)(sc * p0.x); a[1]  = (_Float16)(sc * p0.y);
        a[2]  = (_Float16)(sc * p0.z); a[3]  = (_Float16)(sc * p0.w);
        a[4]  = (_Float16)(sc * p1.x); a[5]  = (_Float16)(sc * p1.y);
        a[6]  = (_Float16)(sc * p1.z); a[7]  = (_Float16)(sc * p1.w);
        a[8]  = (_Float16)(sc * p2.x); a[9]  = (_Float16)(sc * p2.y);
        a[10] = (_Float16)(sc * p2.z); a[11] = (_Float16)(sc * p2.w);
        a[12] = (_Float16)(sc * p3.x); a[13] = (_Float16)(sc * p3.y);
        a[14] = (_Float16)(sc * p3.z); a[15] = (_Float16)(sc * p3.w);

        // B fragments for all four N-tiles from LDS:
        //   W[col][32*kt + 8*hi + j] (j<8), W[col][32*kt + 16 + 8*hi + j-8] (j>=8).
        const int koff = kt * 32 + hi * 8;
        #pragma unroll
        for (int nt = 0; nt < 4; ++nt) {
            const _Float16* wp = &sW[(nt * 16 + nc) * 320 + koff];
            const v8h bb0 = *(const v8h*)(wp);
            const v8h bb1 = *(const v8h*)(wp + 16);
            const v16h b = __builtin_shufflevector(bb0, bb1,
                                                   0, 1, 2, 3, 4, 5, 6, 7,
                                                   8, 9, 10, 11, 12, 13, 14, 15);
            if (nt == 0)
                c0 = __builtin_amdgcn_wmma_f32_16x16x32_f16(false, a, false, b,
                                                            (short)0, c0, false, false);
            else if (nt == 1)
                c1 = __builtin_amdgcn_wmma_f32_16x16x32_f16(false, a, false, b,
                                                            (short)0, c1, false, false);
            else if (nt == 2)
                c2 = __builtin_amdgcn_wmma_f32_16x16x32_f16(false, a, false, b,
                                                            (short)0, c2, false, false);
            else
                c3 = __builtin_amdgcn_wmma_f32_16x16x32_f16(false, a, false, b,
                                                            (short)0, c3, false, false);
        }
    }

    // Epilogue: bias + tanh, write 16x64 tile owned by this wave.
    #pragma unroll
    for (int nt = 0; nt < 4; ++nt) {
        const int col = nt * 16 + nc;
        const float bv = bias[col];
        const v8f& c = (nt == 0) ? c0 : (nt == 1) ? c1 : (nt == 2) ? c2 : c3;
        #pragma unroll
        for (int v = 0; v < 8; ++v) {
            const int r = mbase + hi * 8 + v;
            if (r < n) out[(size_t)r * FDIM + col] = tanhf(c[v] + bv);
        }
    }
}

// ---------------------------------------------------------------------------
// Host-side orchestration (graph-capture safe: async ops on `stream` only).
// Workspace layout (floats): 8 buffers of n*64, then att_e[4*b], att_f[4*b].
// ---------------------------------------------------------------------------
extern "C" void kernel_launch(void* const* d_in, const int* in_sizes, int n_in,
                              void* d_out, int out_size, void* d_ws, size_t ws_size,
                              hipStream_t stream)
{
    const float* e     = (const float*)d_in[0];
    const float* f     = (const float*)d_in[1];
    const int*   rowsG = (const int*)d_in[2];
    const int*   colsG = (const int*)d_in[3];
    const float* valsG = (const float*)d_in[4];
    const int*   rowsB = (const int*)d_in[5];
    const int*   colsB = (const int*)d_in[6];
    const float* valsB = (const float*)d_in[7];
    const int*   rows1 = (const int*)d_in[8];
    const int*   cols1 = (const int*)d_in[9];
    const float* vals1 = (const float*)d_in[10];
    const int*   rows2 = (const int*)d_in[11];
    const int*   cols2 = (const int*)d_in[12];
    const float* vals2 = (const float*)d_in[13];
    const float* G_d   = (const float*)d_in[14];
    const float* B_d   = (const float*)d_in[15];
    const float* Pd    = (const float*)d_in[16];
    const float* Qd    = (const float*)d_in[17];
    const float* W_v1  = (const float*)d_in[18];
    const float* b_v1  = (const float*)d_in[19];
    const float* W_v2  = (const float*)d_in[20];
    const float* b_v2  = (const float*)d_in[21];
    const float* w_ae  = (const float*)d_in[22];
    const float* b_ae  = (const float*)d_in[23];
    const float* w_af  = (const float*)d_in[24];
    const float* b_af  = (const float*)d_in[25];

    const int n       = in_sizes[0] / FDIM;
    const int nGraphs = n / NODES;
    const int nnz     = in_sizes[2];
    const size_t nF   = (size_t)n * FDIM;

    float* bufs  = (float*)d_ws;
    float* b0 = bufs + 0 * nF;   // e_G -> e3
    float* b1 = bufs + 1 * nF;   // f_G -> new_e
    float* b2 = bufs + 2 * nF;   // e_B -> f3
    float* b3 = bufs + 3 * nF;   // f_B -> new_f
    float* b4 = bufs + 4 * nF;   // e1
    float* b5 = bufs + 5 * nF;   // f1
    float* b6 = bufs + 6 * nF;   // e2
    float* b7 = bufs + 7 * nF;   // f2
    float* att_e = bufs + 8 * nF;
    float* att_f = att_e + 4 * nGraphs;

    float* out_e = (float*)d_out;
    float* out_f = out_e + nF;

    // 1) Zero SPMM accumulators (graph-capturable async memset).
    hipMemsetAsync(d_ws, 0, 8 * nF * sizeof(float), stream);

    // 2) SPMMs (each fuses e and f for one edge set).
    const int gridS = (nnz + 7) / 8;
    spmm2_kernel<<<gridS, 256, 0, stream>>>(rowsG, colsG, valsG, e, f, b0, b1, nnz);
    spmm2_kernel<<<gridS, 256, 0, stream>>>(rowsB, colsB, valsB, e, f, b2, b3, nnz);
    spmm2_kernel<<<gridS, 256, 0, stream>>>(rows1, cols1, vals1, e, f, b4, b5, nnz);
    spmm2_kernel<<<gridS, 256, 0, stream>>>(rows2, cols2, vals2, e, f, b6, b7, nnz);

    // 3) Elementwise algebra (in place over b0..b3).
    const long total = (long)nF;
    const int gridE = (int)((total + 255) / 256);
    elemwise_kernel<<<gridE, 256, 0, stream>>>(e, f, b0, b1, b2, b3,
                                               G_d, B_d, Pd, Qd, total);

    // 4) Attention scalars per graph.
    attention_kernel<<<nGraphs, 256, 0, stream>>>(b0, b1, b4, b6,
                                                  b2, b3, b5, b7,
                                                  w_ae, b_ae, w_af, b_af,
                                                  att_e, att_f, nGraphs);

    // 5) Fused concat-GEMM + bias + tanh (WMMA), e side then f side.
    //    8 waves/block, 128 rows/block.
    const int gridM = (n + 127) / 128;
    gemm_att_tanh_kernel<<<gridM, 256, 0, stream>>>(b0, b1, b4, b6, e, att_e,
                                                    W_v1, b_v1, out_e, n, nGraphs);
    gemm_att_tanh_kernel<<<gridM, 256, 0, stream>>>(b2, b3, b5, b7, f, att_f,
                                                    W_v2, b_v2, out_f, n, nGraphs);
}
